// ViTMultiHeadAttention_49692771615029
// MI455X (gfx1250) — matrix operand order repaired and checked by reading, not compiled, not run on hardware
//
#include <hip/hip_runtime.h>
#include <hip/hip_bf16.h>
#include <stdint.h>

// Problem constants (match reference)
#define BB   8
#define TT   1024
#define DIMM 1024
#define HH   16
#define DH   64
#define N3   (3 * DIMM)   // 3072
#define MM   (BB * TT)    // 8192

typedef __attribute__((ext_vector_type(16))) __bf16 bf16x16;
typedef __attribute__((ext_vector_type(8)))  float  f32x8;
typedef __attribute__((ext_vector_type(4)))  unsigned int u32x4;
typedef __attribute__((ext_vector_type(8)))  int    i32x8;
typedef __attribute__((ext_vector_type(4)))  int    i32x4;

union Frag {
    bf16x16 v;
    uint4   q[2];
};

static __device__ __forceinline__ __bf16 f2bf(float x) {
    return (__bf16)x;   // RNE fptrunc -> v_cvt on gfx1250
}

static __device__ __forceinline__ f32x8 wmma_bf16(bf16x16 a, bf16x16 b, f32x8 c) {
    // D = A(16x32 bf16) * B(32x16 bf16) + C(16x16 f32)
    return __builtin_amdgcn_wmma_f32_16x16x32_bf16(
        /*neg_a=*/false, a, /*neg_b=*/false, b,
        /*c_mod=*/(short)0, c, /*reuse_a=*/false, /*reuse_b=*/false);
}

// ---------------------------------------------------------------------------
// TDM: load a 2-D tile (tile_d1 rows x tile_d0 elements, 2-byte elements)
// from a row-major tensor (row stride = stride_d0 elements) into LDS.
// Descriptor per CDNA5 ISA ch.8 (D# groups 0/1; groups 2/3 NULL for 2-D).
// amdgpu-toolchain builtin: 6 args (g0, g1, g2, g3, g4, cpol).
// ---------------------------------------------------------------------------
static __device__ __forceinline__ void tdm_load_2d(
    uint32_t lds_addr, const void* gptr,
    uint32_t tensor_d0, uint32_t tensor_d1,
    uint32_t tile_d0, uint32_t tile_d1, uint32_t stride_d0)
{
    uint64_t ga = (uint64_t)(uintptr_t)gptr;
    u32x4 g0;
    g0[0] = 1u;                                   // count=1 (valid), user desc
    g0[1] = lds_addr;                             // LDS byte address
    g0[2] = (uint32_t)ga;                         // global_addr[31:0]
    g0[3] = ((uint32_t)(ga >> 32) & 0x01FFFFFFu)  // global_addr[56:32]
            | (2u << 30);                         // type = 2 ("image")
    i32x8 g1;
    g1[0] = (int)(1u << 16);                      // wg_mask=0, data_size=1 (2B)
    g1[1] = (int)((tensor_d0 & 0xFFFFu) << 16);   // tensor_dim0[15:0] @ [63:48]
    g1[2] = (int)((tensor_d0 >> 16) | ((tensor_d1 & 0xFFFFu) << 16));
    g1[3] = (int)((tensor_d1 >> 16) | (tile_d0 << 16));   // tile_dim0 @ [127:112]
    g1[4] = (int)tile_d1;                         // tile_dim1; tile_dim2=0
    g1[5] = (int)stride_d0;                       // tensor_dim0_stride[31:0]
    g1[6] = 0;                                    // stride hi / dim1_stride lo
    g1[7] = 0;
    i32x4 z4 = {0, 0, 0, 0};
    i32x8 z8 = {0, 0, 0, 0, 0, 0, 0, 0};
    __builtin_amdgcn_tensor_load_to_lds(g0, g1, z4, z4, z8, 0);
}

// ---------------------------------------------------------------------------
// Kernel 1: fp32 -> bf16 conversion, 4 elements per thread
// ---------------------------------------------------------------------------
__global__ __launch_bounds__(256)
void cvt_f32_bf16(const float* __restrict__ in, __bf16* __restrict__ out, int n4) {
    int i = blockIdx.x * blockDim.x + threadIdx.x;
    if (i >= n4) return;
    float4 f = ((const float4*)in)[i];
    union { __bf16 b[4]; uint2 u; } r;
    r.b[0] = f2bf(f.x); r.b[1] = f2bf(f.y);
    r.b[2] = f2bf(f.z); r.b[3] = f2bf(f.w);
    ((uint2*)out)[i] = r.u;
}

// ---------------------------------------------------------------------------
// Kernel 2: QKV projection GEMM.  C[M,N3] = A[M,K] * W[N3,K]^T + bias, bf16 out
// Block = 256 thr = 8 waves (2x4), block tile 128x256, wave tile 64x64.
// A/B tiles staged through LDS by the Tensor Data Mover, double-buffered.
// ---------------------------------------------------------------------------
__global__ __launch_bounds__(256)
void qkv_gemm(const __bf16* __restrict__ A,   // [M, K] bf16 (x)
              const __bf16* __restrict__ W,   // [N3, K] bf16 (W_qkv, k contiguous)
              const float*  __restrict__ bias,// [N3]
              __bf16* __restrict__ C)         // [M, N3] bf16
{
    __shared__ __align__(16) __bf16 ldsA[2][128 * 32];   // 2 x 8 KB
    __shared__ __align__(16) __bf16 ldsB[2][256 * 32];   // 2 x 16 KB

    const int lane  = threadIdx.x & 31;
    const int wave  = threadIdx.x >> 5;
    const int waveM = wave >> 2;           // 0..1
    const int waveN = wave & 3;            // 0..3
    const int blockM = blockIdx.x * 128;
    const int blockN = blockIdx.y * 256;

    const int mrow   = lane & 15;          // A row / B col within fragment
    const int hi     = lane >> 4;
    const int khalfA = hi * 8;             // A-frag K offset of lane half
    const int koffB  = hi * 8;             // B-frag K offset of lane half

    f32x8 acc[4][4] = {};

    // Prologue: TDM-stage k0=0 tiles into buffer 0 (TDM ignores EXEC; one
    // issue per workgroup from wave 0).
    if (wave == 0) {
        tdm_load_2d((uint32_t)(uintptr_t)&ldsA[0][0], A + (size_t)blockM * DIMM,
                    DIMM, MM, 32, 128, DIMM);
        tdm_load_2d((uint32_t)(uintptr_t)&ldsB[0][0], W + (size_t)blockN * DIMM,
                    DIMM, N3, 32, 256, DIMM);
    }

    for (int k0 = 0; k0 < DIMM; k0 += 32) {
        const int cur = (k0 >> 5) & 1;
        const int nxt = cur ^ 1;

        __syncthreads();   // all waves finished reading buf[nxt] (prev cur)
        if (wave == 0) {
            if (k0 + 32 < DIMM) {
                tdm_load_2d((uint32_t)(uintptr_t)&ldsA[nxt][0],
                            A + (size_t)blockM * DIMM + (k0 + 32),
                            DIMM, MM, 32, 128, DIMM);
                tdm_load_2d((uint32_t)(uintptr_t)&ldsB[nxt][0],
                            W + (size_t)blockN * DIMM + (k0 + 32),
                            DIMM, N3, 32, 256, DIMM);
                // in-order completion: <=2 outstanding => buf[cur] resident
                __builtin_amdgcn_s_wait_tensorcnt((short)2);
            } else {
                __builtin_amdgcn_s_wait_tensorcnt((short)0);
            }
        }
        __syncthreads();   // buf[cur] visible to all waves

        Frag a[4], b[4];
#pragma unroll
        for (int i = 0; i < 4; ++i) {
            const __bf16* pa = &ldsA[cur][(waveM * 64 + i * 16 + mrow) * 32 + khalfA];
            a[i].q[0] = *(const uint4*)(pa);        // K = khalfA .. +7
            a[i].q[1] = *(const uint4*)(pa + 16);   // K = khalfA+16 .. +23
        }
#pragma unroll
        for (int j = 0; j < 4; ++j) {
            const __bf16* pb = &ldsB[cur][(waveN * 64 + j * 16 + mrow) * 32 + koffB];
            b[j].q[0] = *(const uint4*)(pb);        // K = koffB .. +7
            b[j].q[1] = *(const uint4*)(pb + 16);   // K = koffB+16 .. +23
        }
#pragma unroll
        for (int i = 0; i < 4; ++i)
#pragma unroll
            for (int j = 0; j < 4; ++j)
                acc[i][j] = wmma_bf16(a[i].v, b[j].v, acc[i][j]);
    }

    // Epilogue: +bias, f32 -> bf16, store. C layout: M = r + 8*hi, N = lane&15.
#pragma unroll
    for (int j = 0; j < 4; ++j) {
        const int col = blockN + waveN * 64 + j * 16 + mrow;
        const float bv = bias[col];
#pragma unroll
        for (int i = 0; i < 4; ++i) {
#pragma unroll
            for (int r = 0; r < 8; ++r) {
                const int row = blockM + waveM * 64 + i * 16 + r + 8 * hi;
                C[(size_t)row * N3 + col] = f2bf(acc[i][j][r] + bv);
            }
        }
    }
}

// ---------------------------------------------------------------------------
// Kernel 3: V transpose  vt[b,h,d,t] = qkv[b,t, V, h, d]
// ---------------------------------------------------------------------------
__global__ __launch_bounds__(256)
void v_transpose(const __bf16* __restrict__ qkv, __bf16* __restrict__ vt) {
    int idx = blockIdx.x * blockDim.x + threadIdx.x;   // over B*H*DH*T = 8M
    int t = idx & (TT - 1);
    int d = (idx >> 10) & (DH - 1);
    int h = (idx >> 16) & (HH - 1);
    int b = idx >> 20;
    vt[idx] = qkv[(size_t)(b * TT + t) * N3 + 2 * DIMM + h * DH + d];
}

// ---------------------------------------------------------------------------
// Kernel 4: flash attention. One wave per 16-row query tile; kv blocks of 32.
// ---------------------------------------------------------------------------
__global__ __launch_bounds__(256)
void flash_attn(const __bf16* __restrict__ qkv,  // [B,T,3,H,DH] bf16
                const __bf16* __restrict__ vt,   // [B,H,DH,T]   bf16
                float* __restrict__ out)         // [B,T,DIM]    f32
{
    const int lane = threadIdx.x & 31;
    const int wave = threadIdx.x >> 5;
    const int gw = blockIdx.x * 8 + wave;       // 0..8191
    const int qt = gw & 63;                     // query tile within (b,h)
    const int h  = (gw >> 6) & (HH - 1);
    const int b  = gw >> 10;

    __shared__ __align__(16) __bf16 psh[8][16 * 32];
    __bf16* pbuf = psh[wave];

    const int mrow   = lane & 15;
    const int hi     = lane >> 4;
    const int khalfA = hi * 8;
    const int koffB  = hi * 8;

    // Q fragments (two 32-wide k-steps over d), A-layout
    Frag aq[2];
    {
        const __bf16* q = qkv + (size_t)(b * TT + qt * 16 + mrow) * N3 + h * DH;
#pragma unroll
        for (int kk = 0; kk < 2; ++kk) {
            const __bf16* p = q + kk * 32 + khalfA;
            aq[kk].q[0] = *(const uint4*)(p);
            aq[kk].q[1] = *(const uint4*)(p + 16);
        }
    }

    f32x8 o[4] = {};
    float m[8], l[8];
#pragma unroll
    for (int r = 0; r < 8; ++r) { m[r] = -1e30f; l[r] = 0.0f; }

    const __bf16* kbase = qkv + (size_t)b * TT * N3 + DIMM + h * DH;     // K chunk
    const __bf16* vbase = vt + (size_t)((b * HH + h) * DH) * TT;         // [d][t]
    const float scale = 0.03125f;   // 1/sqrt(DIM) = 1/32 (reference uses DIM)

    for (int j0 = 0; j0 < TT; j0 += 32) {
        // ---- S = Q K^T for two 16-col tiles -------------------------------
        f32x8 s[2] = {};
#pragma unroll
        for (int sn = 0; sn < 2; ++sn) {
            const __bf16* kr = kbase + (size_t)(j0 + sn * 16 + mrow) * N3;
#pragma unroll
            for (int kk = 0; kk < 2; ++kk) {
                Frag bk;
                const __bf16* p = kr + kk * 32 + koffB;
                bk.q[0] = *(const uint4*)(p);
                bk.q[1] = *(const uint4*)(p + 16);
                s[sn] = wmma_bf16(aq[kk].v, bk.v, s[sn]);
            }
        }

        // ---- online softmax update (rows live in 16-lane groups) ----------
        float corr[8];
#pragma unroll
        for (int r = 0; r < 8; ++r) {
            s[0][r] *= scale;
            s[1][r] *= scale;
            float v = fmaxf(s[0][r], s[1][r]);
            v = fmaxf(v, __shfl_xor(v, 1));
            v = fmaxf(v, __shfl_xor(v, 2));
            v = fmaxf(v, __shfl_xor(v, 4));
            v = fmaxf(v, __shfl_xor(v, 8));
            const float mnew = fmaxf(m[r], v);
            corr[r] = __expf(m[r] - mnew);
            m[r] = mnew;
        }
#pragma unroll
        for (int r = 0; r < 8; ++r) {
            const float p0 = __expf(s[0][r] - m[r]);
            const float p1 = __expf(s[1][r] - m[r]);
            float rs = p0 + p1;
            rs += __shfl_xor(rs, 1);
            rs += __shfl_xor(rs, 2);
            rs += __shfl_xor(rs, 4);
            rs += __shfl_xor(rs, 8);
            l[r] = l[r] * corr[r] + rs;
            const int row = r + 8 * hi;          // C layout row
            pbuf[row * 32 + mrow]      = f2bf(p0);
            pbuf[row * 32 + 16 + mrow] = f2bf(p1);
        }
#pragma unroll
        for (int dt = 0; dt < 4; ++dt)
#pragma unroll
            for (int r = 0; r < 8; ++r)
                o[dt][r] *= corr[r];

        // ---- reload P from LDS in A-fragment layout -----------------------
        Frag pa;
        {
            const __bf16* pp = pbuf + mrow * 32 + khalfA;
            pa.q[0] = *(const uint4*)(pp);
            pa.q[1] = *(const uint4*)(pp + 16);
        }

        // ---- O += P V (4 d-tiles of 16) -----------------------------------
#pragma unroll
        for (int dt = 0; dt < 4; ++dt) {
            Frag bv;
            const __bf16* vp = vbase + (size_t)(dt * 16 + mrow) * TT + j0 + koffB;
            bv.q[0] = *(const uint4*)(vp);
            bv.q[1] = *(const uint4*)(vp + 16);
            o[dt] = wmma_bf16(pa.v, bv.v, o[dt]);
        }
    }

    // ---- normalize and store (f32, coalesced across the 16-lane group) ----
#pragma unroll
    for (int r = 0; r < 8; ++r) {
        const float inv = 1.0f / l[r];
        const int t = qt * 16 + r + 8 * hi;
        float* op = out + (size_t)(b * TT + t) * DIMM + h * DH;
#pragma unroll
        for (int dt = 0; dt < 4; ++dt)
            op[dt * 16 + mrow] = o[dt][r] * inv;
    }
}

// ---------------------------------------------------------------------------
extern "C" void kernel_launch(void* const* d_in, const int* in_sizes, int n_in,
                              void* d_out, int out_size, void* d_ws, size_t ws_size,
                              hipStream_t stream) {
    const float* x    = (const float*)d_in[0];   // [8,1024,1024]
    const float* Wq   = (const float*)d_in[1];   // [3072,1024]
    const float* bias = (const float*)d_in[2];   // [3072]
    float* out = (float*)d_out;

    // Workspace layout (bf16 buffers), ~86 MB total
    char* ws = (char*)d_ws;
    __bf16* xb   = (__bf16*)ws;                                  // 16 MB
    __bf16* wb   = (__bf16*)(ws + (size_t)MM * DIMM * 2);        //  6 MB
    __bf16* qkvb = (__bf16*)(ws + (size_t)(MM + N3) * DIMM * 2); // 48 MB
    __bf16* vtb  = (__bf16*)(ws + (size_t)(MM + N3) * DIMM * 2
                                + (size_t)MM * N3 * 2);          // 16 MB

    // 1) convert x and W to bf16
    {
        int n4 = MM * DIMM / 4;   // 2,097,152
        cvt_f32_bf16<<<n4 / 256, 256, 0, stream>>>(x, xb, n4);
        int w4 = N3 * DIMM / 4;   // 786,432
        cvt_f32_bf16<<<w4 / 256, 256, 0, stream>>>(Wq, wb, w4);
    }
    // 2) QKV projection (TDM-staged LDS tiles)
    {
        dim3 grid(MM / 128, N3 / 256);   // (64, 12)
        qkv_gemm<<<grid, 256, 0, stream>>>(xb, wb, bias, qkvb);
    }
    // 3) V transpose
    {
        int n = BB * HH * DH * TT;       // 8,388,608
        v_transpose<<<n / 256, 256, 0, stream>>>(qkvb, vtb);
    }
    // 4) flash attention
    {
        int waves = BB * HH * (TT / 16); // 8192 query tiles
        flash_attn<<<waves / 8, 256, 0, stream>>>(qkvb, vtb, out);
    }
    (void)in_sizes; (void)n_in; (void)out_size; (void)ws_size;
}
